// DSS_5772436046424
// MI455X (gfx1250) — matrix operand order repaired and checked by reading, not compiled, not run on hardware
//
#include <hip/hip_runtime.h>
#include <cstddef>
#include <cstdint>

// ---------------------------------------------------------------------------
// DSS (S4-style diagonal state space) forward for MI455X (gfx1250, wave32).
//
// y[b,l,d] = residual + sum_{m<=l} K[m,d] * u[b,l-m,d]
//   K[m,d] = Re( sum_n C'[d,n] e^{Lam_n m} ),  u = LayerNorm(x)
//
// Chunked state-passing algorithm (Q=128, NCH=32 chunks):
//   local  : per-d lower-triangular Toeplitz(K[0:Q,d]) @ u_chunk   (WMMA bf16)
//   collect: A[n,d]   = sum_s e^{Lam (Q-1-s)} u[s,d]               (WMMA bf16)
//   state  : s <- e^{Lam Q} s + A                                  (VALU)
//   emit   : y[t,d]  += Re( sum_n e^{Lam (t+1)} s[n,d] C'[d,n] )   (WMMA bf16)
//
// GEMMs: v_wmma_f32_16x16x32_bf16, f32 accumulation. All fragment loads use
// uniform-base + 32-bit offsets (saddr form). Toeplitz kernel stages K^T rows
// into LDS via the Tensor Data Mover (tensor_load_to_lds + s_wait_tensorcnt).
// Workspace requirement: ~112 MB.
// ---------------------------------------------------------------------------

typedef __bf16 bf16;
typedef __attribute__((ext_vector_type(16))) __bf16 v16bf;
typedef __attribute__((ext_vector_type(8)))  float  v8f;
typedef __attribute__((ext_vector_type(4)))  float  f32x4;
typedef __attribute__((ext_vector_type(4)))  unsigned int u32x4;
typedef __attribute__((ext_vector_type(8)))  int    i32x8;
typedef __attribute__((ext_vector_type(4)))  int    i32x4;

#define WMMA_BF16(a, b, c) \
  __builtin_amdgcn_wmma_f32_16x16x32_bf16(false, (a), false, (b), (short)0, (c), false, false)

// Problem constants
#define BB   4
#define LL   4096
#define DD   1024
#define NN   512
#define QQ   128
#define NCH  32
#define BL_  ((size_t)BB * LL)
#define BLI  (BB * LL)

// ----------------------------- fragment loaders -----------------------------
// A-matrix 16x32 bf16, row-major [m][k] (k contiguous). Lanes 0-15
// hold row m, K 0..15 (v0-7); lanes 16-31 hold K 16..31 (matches ldfragB).
// Offsets are 32-bit so the compiler can use saddr + voffset addressing.
__device__ __forceinline__ v16bf ldfragA(const bf16* __restrict__ base, int lda,
                                         int m0, int k0) {
  int lane = threadIdx.x & 31;
  int off = (m0 + (lane & 15)) * lda + k0 + ((lane >> 4) << 4);
  union { v16bf v; f32x4 h[2]; } u;
  u.h[0] = *(const f32x4*)(base + off);
  u.h[1] = *(const f32x4*)(base + off + 8);
  return u.v;
}

// B-matrix 32x16 bf16 stored TRANSPOSED: Bt[n][k], k contiguous. Same k order as ldfragA:
// lanes 0-15 hold column n, K 0..15 (v0-7); lanes 16-31 hold K 16..31.
__device__ __forceinline__ v16bf ldfragB(const bf16* __restrict__ base, int ldb,
                                         int n0, int k0) {
  int lane = threadIdx.x & 31;
  int off = (n0 + (lane & 15)) * ldb + k0 + ((lane >> 4) << 4);
  union { v16bf v; f32x4 h[2]; } u;
  u.h[0] = *(const f32x4*)(base + off);
  u.h[1] = *(const f32x4*)(base + off + 8);
  return u.v;
}

__device__ __forceinline__ void zero_acc(v8f acc[2][4]) {
#pragma unroll
  for (int i = 0; i < 2; ++i)
#pragma unroll
    for (int j = 0; j < 4; ++j)
#pragma unroll
      for (int r = 0; r < 8; ++r) acc[i][j][r] = 0.f;
}

// Wave computes a 32(M) x 64(N) output tile. out = sum_k A1*B1 (+ A2*B2).
template <int NPROD>
__device__ __forceinline__ void gemm_wave_core(const bf16* __restrict__ A1,
                                               const bf16* __restrict__ A2, int lda,
                                               const bf16* __restrict__ B1,
                                               const bf16* __restrict__ B2, int ldb,
                                               int m0, int n0, int K, v8f acc[2][4]) {
  for (int k0 = 0; k0 < K; k0 += 32) {
    v16bf a0 = ldfragA(A1, lda, m0, k0);
    v16bf a1 = ldfragA(A1, lda, m0 + 16, k0);
    v16bf b0 = ldfragB(B1, ldb, n0, k0);
    v16bf b1 = ldfragB(B1, ldb, n0 + 16, k0);
    v16bf b2 = ldfragB(B1, ldb, n0 + 32, k0);
    v16bf b3 = ldfragB(B1, ldb, n0 + 48, k0);
    acc[0][0] = WMMA_BF16(a0, b0, acc[0][0]);
    acc[0][1] = WMMA_BF16(a0, b1, acc[0][1]);
    acc[0][2] = WMMA_BF16(a0, b2, acc[0][2]);
    acc[0][3] = WMMA_BF16(a0, b3, acc[0][3]);
    acc[1][0] = WMMA_BF16(a1, b0, acc[1][0]);
    acc[1][1] = WMMA_BF16(a1, b1, acc[1][1]);
    acc[1][2] = WMMA_BF16(a1, b2, acc[1][2]);
    acc[1][3] = WMMA_BF16(a1, b3, acc[1][3]);
    if constexpr (NPROD == 2) {
      v16bf c0 = ldfragA(A2, lda, m0, k0);
      v16bf c1 = ldfragA(A2, lda, m0 + 16, k0);
      v16bf d0 = ldfragB(B2, ldb, n0, k0);
      v16bf d1 = ldfragB(B2, ldb, n0 + 16, k0);
      v16bf d2 = ldfragB(B2, ldb, n0 + 32, k0);
      v16bf d3 = ldfragB(B2, ldb, n0 + 48, k0);
      acc[0][0] = WMMA_BF16(c0, d0, acc[0][0]);
      acc[0][1] = WMMA_BF16(c0, d1, acc[0][1]);
      acc[0][2] = WMMA_BF16(c0, d2, acc[0][2]);
      acc[0][3] = WMMA_BF16(c0, d3, acc[0][3]);
      acc[1][0] = WMMA_BF16(c1, d0, acc[1][0]);
      acc[1][1] = WMMA_BF16(c1, d1, acc[1][1]);
      acc[1][2] = WMMA_BF16(c1, d2, acc[1][2]);
      acc[1][3] = WMMA_BF16(c1, d3, acc[1][3]);
    }
  }
}

// C/D layout: v8f element r -> row m0+r (lanes 0-15) or m0+8+r (lanes 16-31),
// column n0 + (lane&15).
__device__ __forceinline__ void tile_store(float* __restrict__ o, int ldo,
                                           int m0, int n0, v8f a) {
  int lane = threadIdx.x & 31;
  int off = (m0 + ((lane >> 4) << 3)) * ldo + n0 + (lane & 15);
#pragma unroll
  for (int r = 0; r < 8; ++r) o[off + r * ldo] = a[r];
}
__device__ __forceinline__ void tile_add(float* __restrict__ o, int ldo,
                                         int m0, int n0, v8f a) {
  int lane = threadIdx.x & 31;
  int off = (m0 + ((lane >> 4) << 3)) * ldo + n0 + (lane & 15);
#pragma unroll
  for (int r = 0; r < 8; ++r) o[off + r * ldo] += a[r];
}

// --------------------- Tensor Data Mover: 1D row -> LDS ---------------------
// Stages `nelem` contiguous f32 from global into LDS via TDM. Descriptor per
// CDNA5 ISA (2D tensor, tile = 1 row of nelem elems).
// This toolchain exposes the 6-arg builtin (g0, g1, g2, g3, g4, cpol).
__device__ __forceinline__ void tdm_load_row_f32(const float* gaddr,
                                                 unsigned lds_off, int nelem) {
  unsigned long long ga = (unsigned long long)(uintptr_t)gaddr;
  u32x4 g0;
  g0[0] = 1u;                                   // count=1, user descriptor
  g0[1] = lds_off;                              // lds_addr (bytes)
  g0[2] = (unsigned)(ga & 0xFFFFFFFFu);         // global_addr[31:0]
  g0[3] = (unsigned)((ga >> 32) & 0x01FFFFFFu)  // global_addr[56:32]
          | (2u << 30);                         // type = 2 ("image")
  i32x8 g1;
  g1[0] = (2 << 16);                            // data_size = 4 bytes
  g1[1] = (nelem & 0xFFFF) << 16;               // tensor_dim0[15:0]
  g1[2] = ((nelem >> 16) & 0xFFFF)              // tensor_dim0[31:16]
          | (1 << 16);                          // tensor_dim1 = 1 (lo16)
  g1[3] = (nelem & 0xFFFF) << 16;               // tile_dim0 = nelem
  g1[4] = 1;                                    // tile_dim1 = 1, tile_dim2 = 0
  g1[5] = nelem;                                // tensor_dim0_stride lo32
  g1[6] = 0;
  g1[7] = 0;
  i32x4 z4 = {0, 0, 0, 0};
  i32x8 z8 = {0, 0, 0, 0, 0, 0, 0, 0};
  __builtin_amdgcn_tensor_load_to_lds(g0, g1, z4, z4, z8, 0);
}

// ------------------------------- prep kernels -------------------------------
__global__ void dss_prep0(const float* __restrict__ Lr, const float* __restrict__ Li,
                          float* lrA, float* liA, float* Zr, float* Zi,
                          float* wr, float* wi) {
  int n = blockIdx.x * blockDim.x + threadIdx.x;
  if (n >= NN) return;
  float lr = -__expf(Lr[n]);
  float li = __expf(Li[n]);
  lrA[n] = lr; liA[n] = li;
  // w = (e^Lam - 1) / Lam
  float er = __expf(lr); float sn, cs; __sincosf(li, &sn, &cs);
  float exr = er * cs - 1.f, exi = er * sn;
  float den = 1.f / (lr * lr + li * li);
  wr[n] = (exr * lr + exi * li) * den;
  wi[n] = (exi * lr - exr * li) * den;
  // Z = e^{Lam * Q}
  float eQ = __expf(lr * (float)QQ); float snQ, csQ; __sincosf(li * (float)QQ, &snQ, &csQ);
  Zr[n] = eQ * csQ; Zi[n] = eQ * snQ;
}

// C' = (C_real + i C_imag) * w ; stored [D][N] both as f32 (for mkG) and bf16 (GEMM)
__global__ void dss_prepc(const float* __restrict__ Cr, const float* __restrict__ Ci,
                          const float* __restrict__ wr, const float* __restrict__ wi,
                          float* Cpr, float* Cpi, bf16* Cbr, bf16* Cbi) {
  int idx = blockIdx.x * blockDim.x + threadIdx.x;  // D*N
  int n = idx & (NN - 1);
  float cr = Cr[idx], ci = Ci[idx];
  float pr = cr * wr[n] - ci * wi[n];
  float pi = cr * wi[n] + ci * wr[n];
  Cpr[idx] = pr; Cpi[idx] = pi;
  Cbr[idx] = (bf16)pr; Cbi[idx] = (bf16)pi;
}

// KS[t][n]=e^{Lam t}; E[t][n]=e^{Lam (t+1)}; Pcat[n][s]=e^{Lam (Q-1-s)} (real rows
// 0..N-1, imag rows N..2N-1). Imag A/B-operands stored NEGATED (bf16 WMMA lacks A-neg).
__global__ void dss_prepse(const float* __restrict__ lrA, const float* __restrict__ liA,
                           bf16* KSr, bf16* KSin, bf16* Er, bf16* Ein, bf16* Pcat) {
  int idx = blockIdx.x * blockDim.x + threadIdx.x;  // Q*N
  int n = idx & (NN - 1);
  int t = idx >> 9;
  float lr = lrA[n], li = liA[n];
  float e0 = __expf(lr * (float)t); float s0, c0; __sincosf(li * (float)t, &s0, &c0);
  KSr [t * NN + n] = (bf16)( e0 * c0);
  KSin[t * NN + n] = (bf16)(-e0 * s0);
  float e1 = __expf(lr * (float)(t + 1)); float s1, c1; __sincosf(li * (float)(t + 1), &s1, &c1);
  Er [t * NN + n] = (bf16)( e1 * c1);
  Ein[t * NN + n] = (bf16)(-e1 * s1);
  float e2 = __expf(lr * (float)(QQ - 1 - t)); float s2, c2; __sincosf(li * (float)(QQ - 1 - t), &s2, &c2);
  Pcat[n * QQ + t]        = (bf16)(e2 * c2);
  Pcat[(NN + n) * QQ + t] = (bf16)(e2 * s2);
}

// ----------------------- LayerNorm + residual + bf16 u ----------------------
__global__ void dss_ln(const float* __restrict__ x, const float* __restrict__ lw,
                       const float* __restrict__ lb, const float* __restrict__ pD,
                       float* __restrict__ out, bf16* __restrict__ uB) {
  int row = blockIdx.x;  // b*L + l
  const float* xr = x + (size_t)row * DD;
  float v[4], s = 0.f, s2 = 0.f;
#pragma unroll
  for (int i = 0; i < 4; ++i) {
    v[i] = xr[threadIdx.x + i * 256];
    s += v[i]; s2 += v[i] * v[i];
  }
  __shared__ float sm[256], sm2[256];
  sm[threadIdx.x] = s; sm2[threadIdx.x] = s2;
  __syncthreads();
  for (int st = 128; st > 0; st >>= 1) {
    if (threadIdx.x < st) { sm[threadIdx.x] += sm[threadIdx.x + st]; sm2[threadIdx.x] += sm2[threadIdx.x + st]; }
    __syncthreads();
  }
  float mean = sm[0] * (1.f / DD);
  float var  = sm2[0] * (1.f / DD) - mean * mean;
  float rs = rsqrtf(var + 1e-5f);
#pragma unroll
  for (int i = 0; i < 4; ++i) {
    int d = threadIdx.x + i * 256;
    float u = (v[i] - mean) * rs * lw[d] + lb[d];
    out[(size_t)row * DD + d] = u * pD[d];   // residual
    uB [(size_t)row * DD + d] = (bf16)u;
  }
}

// -------------------- transpose u [BL][D] -> uT [D][BL] ---------------------
__global__ void dss_transpose(const bf16* __restrict__ in, bf16* __restrict__ outT) {
  __shared__ bf16 t[32][33];
  int ct = blockIdx.x, rt = blockIdx.y;
  int tx = threadIdx.x, ty = threadIdx.y;
#pragma unroll
  for (int i = 0; i < 32; i += 8)
    t[ty + i][tx] = in[(size_t)(rt * 32 + ty + i) * DD + ct * 32 + tx];
  __syncthreads();
#pragma unroll
  for (int i = 0; i < 32; i += 8)
    outT[(size_t)(ct * 32 + ty + i) * BL_ + rt * 32 + tx] = t[tx][ty + i];
}

// -------- K^T[d][0:Q] = sum_n C'r[d,n] KSr[t,n] + C'i[d,n] (-KSi[t,n]) ------
__global__ void dss_kcon(const bf16* __restrict__ Cbr, const bf16* __restrict__ Cbi,
                         const bf16* __restrict__ KSr, const bf16* __restrict__ KSin,
                         float* __restrict__ KfT) {
  int bid = blockIdx.x;             // 32 m-tiles (d) * 2 n-tiles (tau)
  int m0 = (bid & 31) * 32, n0 = (bid >> 5) * 64;
  v8f acc[2][4]; zero_acc(acc);
  gemm_wave_core<2>(Cbr, Cbi, NN, KSr, KSin, NN, m0, n0, NN, acc);
#pragma unroll
  for (int i = 0; i < 2; ++i)
#pragma unroll
    for (int j = 0; j < 4; ++j)
      tile_store(KfT, QQ, m0 + 16 * i, n0 + 16 * j, acc[i][j]);
}

// ------------- within-chunk causal Toeplitz conv, per channel d -------------
// y[b, c*Q+t, d] += sum_{s<=t} K[t-s,d] u[b, c*Q+s, d] for all (b, c, t).
// K^T row d staged into LDS by the Tensor Data Mover.
__global__ void dss_toeplitz(const float* __restrict__ KfT, const bf16* __restrict__ uT,
                             float* __restrict__ out) {
  const int d = blockIdx.x;
  __shared__ __attribute__((aligned(16))) float Kcol[QQ];
  __shared__ __attribute__((aligned(16))) bf16 T[QQ * 136];  // rows padded to 136

  // Wave 0 issues the TDM transfer (TDM ignores EXEC -> scalar-uniform branch).
  if (__builtin_amdgcn_readfirstlane(threadIdx.x) < 32) {
    tdm_load_row_f32(KfT + (size_t)d * QQ, (unsigned)(uintptr_t)&Kcol[0], QQ);
    __builtin_amdgcn_s_wait_tensorcnt(0);
  }
  __syncthreads();

  for (int i = threadIdx.x; i < QQ * 136; i += blockDim.x) {
    int t = i / 136, s = i - t * 136;
    float v = (s < QQ && s <= t) ? Kcol[t - s] : 0.f;
    T[i] = (bf16)v;
  }
  __syncthreads();

  const bf16* uTd = uT + (size_t)d * BL_;   // uniform base, 32-bit offsets below
  int w = threadIdx.x >> 5;
  int lane = threadIdx.x & 31;
  for (int wt = w; wt < 8; wt += 4) {       // 4 m-tiles x 2 n-tiles of 32x64
    int m0 = (wt & 3) * 32, n0 = (wt >> 2) * 64;
    v8f acc[2][4]; zero_acc(acc);
    for (int k0 = 0; k0 < QQ; k0 += 32) {
      v16bf a0 = ldfragA(T, 136, m0, k0);
      v16bf a1 = ldfragA(T, 136, m0 + 16, k0);
      v16bf b[4];
#pragma unroll
      for (int jj = 0; jj < 4; ++jj) {
        int j  = n0 + 16 * jj + (lane & 15);      // column = (b, chunk) pair
        int off = (j >> 5) * LL + (j & 31) * QQ + k0 + ((lane >> 4) << 4);
        union { v16bf v; f32x4 h[2]; } u_;
        u_.h[0] = *(const f32x4*)(uTd + off);
        u_.h[1] = *(const f32x4*)(uTd + off + 8);
        b[jj] = u_.v;
      }
#pragma unroll
      for (int jj = 0; jj < 4; ++jj) {
        acc[0][jj] = WMMA_BF16(a0, b[jj], acc[0][jj]);
        acc[1][jj] = WMMA_BF16(a1, b[jj], acc[1][jj]);
      }
    }
#pragma unroll
    for (int i2 = 0; i2 < 2; ++i2)
#pragma unroll
      for (int j2 = 0; j2 < 4; ++j2) {
        int j  = n0 + 16 * j2 + (lane & 15);
        int mb = m0 + 16 * i2 + ((lane >> 4) << 3);
        int rowbase = (j >> 5) * LL + (j & 31) * QQ;
        v8f a = acc[i2][j2];
#pragma unroll
        for (int r = 0; r < 8; ++r)
          out[(size_t)(rowbase + mb + r) * DD + d] += a[r];
      }
  }
}

// --------- collect: A[b][m][d] = sum_s Pcat[m][s] u[b, c*Q+s, d] ------------
__global__ void dss_collect(const bf16* __restrict__ Pcat, const bf16* __restrict__ uT,
                            float* __restrict__ Abuf, int c) {
  int bid = blockIdx.x;                       // 32 m * 16 n * B
  int m0 = (bid & 31) * 32, n0 = ((bid >> 5) & 15) * 64, b = bid >> 9;
  const bf16* Bt = uT + b * LL + c * QQ;      // row d stride = BL
  // Warm L2 for the next chunk's u tile (global_prefetch_b8).
  if (c + 1 < NCH) {
    __builtin_prefetch(Bt + QQ + (n0 + (threadIdx.x & 31)) * BLI, 0, 1);
    __builtin_prefetch(Bt + QQ + (n0 + 32 + (threadIdx.x & 31)) * BLI, 0, 1);
  }
  v8f acc[2][4]; zero_acc(acc);
  gemm_wave_core<1>(Pcat, nullptr, QQ, Bt, nullptr, BLI, m0, n0, QQ, acc);
  float* o = Abuf + (size_t)b * (2 * NN) * DD;
#pragma unroll
  for (int i = 0; i < 2; ++i)
#pragma unroll
    for (int j = 0; j < 4; ++j)
      tile_store(o, DD, m0 + 16 * i, n0 + 16 * j, acc[i][j]);
}

// ----------------------- state update: s <- Z*s + A -------------------------
__global__ void dss_update(float* __restrict__ sbuf, const float* __restrict__ Abuf,
                           const float* __restrict__ Zr, const float* __restrict__ Zi) {
  int idx = blockIdx.x * blockDim.x + threadIdx.x;  // B*N*D
  int d = idx & (DD - 1);
  int n = (idx >> 10) & (NN - 1);
  int b = idx >> 19;
  size_t o  = (size_t)b * (2 * NN) * DD + (size_t)n * DD + d;
  size_t oi = o + (size_t)NN * DD;
  float sr = sbuf[o], si = sbuf[oi];
  float ar = Abuf[o], ai = Abuf[oi];
  float zr = Zr[n], zi = Zi[n];
  sbuf[o]  = zr * sr - zi * si + ar;
  sbuf[oi] = zr * si + zi * sr + ai;
}

// ------------------ G[b][d][n] = s[b][n][d] * C'[d][n] ----------------------
__global__ void dss_mkg(const float* __restrict__ sbuf, const float* __restrict__ Cpr,
                        const float* __restrict__ Cpi, bf16* __restrict__ Gr,
                        bf16* __restrict__ Gi) {
  int idx = blockIdx.x * blockDim.x + threadIdx.x;  // B*D*N, n fastest
  int n = idx & (NN - 1);
  int d = (idx >> 9) & (DD - 1);
  int b = idx >> 19;
  const float* sb = sbuf + (size_t)b * (2 * NN) * DD;
  float sr = sb[n * DD + d];
  float si = sb[(NN + n) * DD + d];
  float cr = Cpr[d * NN + n], ci = Cpi[d * NN + n];
  size_t go = (size_t)b * DD * NN + (size_t)d * NN + n;
  Gr[go] = (bf16)(sr * cr - si * ci);
  Gi[go] = (bf16)(sr * ci + si * cr);
}

// ---- emit: y[b, c*Q+t, d] += sum_n Er[t][n] Gr[d][n] - Ei[t][n] Gi[d][n] ---
__global__ void dss_emit(const bf16* __restrict__ Er, const bf16* __restrict__ Ein,
                         const bf16* __restrict__ Gr, const bf16* __restrict__ Gi,
                         float* __restrict__ out, int c) {
  int bid = blockIdx.x;                       // 4 m * 16 n * B
  int m0 = (bid & 3) * 32, n0 = ((bid >> 2) & 15) * 64, b = bid >> 6;
  const bf16* B1 = Gr + (size_t)b * DD * NN;
  const bf16* B2 = Gi + (size_t)b * DD * NN;
  v8f acc[2][4]; zero_acc(acc);
  gemm_wave_core<2>(Er, Ein, NN, B1, B2, NN, m0, n0, NN, acc);
  float* o = out + ((size_t)b * LL + (size_t)c * QQ) * DD;
#pragma unroll
  for (int i = 0; i < 2; ++i)
#pragma unroll
    for (int j = 0; j < 4; ++j)
      tile_add(o, DD, m0 + 16 * i, n0 + 16 * j, acc[i][j]);
}

// --------------------------------- launch -----------------------------------
extern "C" void kernel_launch(void* const* d_in, const int* in_sizes, int n_in,
                              void* d_out, int out_size, void* d_ws, size_t ws_size,
                              hipStream_t stream) {
  const float* x    = (const float*)d_in[0];
  const float* ln_w = (const float*)d_in[1];
  const float* ln_b = (const float*)d_in[2];
  const float* Lr   = (const float*)d_in[3];
  const float* Li   = (const float*)d_in[4];
  const float* Cr   = (const float*)d_in[5];
  const float* Ci   = (const float*)d_in[6];
  const float* pD   = (const float*)d_in[7];
  float* out = (float*)d_out;

  // workspace carve-up (~112 MB)
  char* w = (char*)d_ws;
  auto carve = [&](size_t bytes) {
    void* p = (void*)w;
    w += (bytes + 255) & ~(size_t)255;
    return p;
  };
  bf16*  uB   = (bf16*) carve(BL_ * DD * sizeof(bf16));        // 32 MB
  bf16*  uT   = (bf16*) carve(BL_ * DD * sizeof(bf16));        // 32 MB
  float* KfT  = (float*)carve((size_t)DD * QQ * sizeof(float));
  bf16*  KSr  = (bf16*) carve((size_t)QQ * NN * sizeof(bf16));
  bf16*  KSin = (bf16*) carve((size_t)QQ * NN * sizeof(bf16));
  bf16*  Er   = (bf16*) carve((size_t)QQ * NN * sizeof(bf16));
  bf16*  Ein  = (bf16*) carve((size_t)QQ * NN * sizeof(bf16));
  bf16*  Pcat = (bf16*) carve((size_t)2 * NN * QQ * sizeof(bf16));
  bf16*  Cbr  = (bf16*) carve((size_t)DD * NN * sizeof(bf16));
  bf16*  Cbi  = (bf16*) carve((size_t)DD * NN * sizeof(bf16));
  float* Cpr  = (float*)carve((size_t)DD * NN * sizeof(float));
  float* Cpi  = (float*)carve((size_t)DD * NN * sizeof(float));
  float* lrA  = (float*)carve(NN * sizeof(float));
  float* liA  = (float*)carve(NN * sizeof(float));
  float* Zr   = (float*)carve(NN * sizeof(float));
  float* Zi   = (float*)carve(NN * sizeof(float));
  float* wr   = (float*)carve(NN * sizeof(float));
  float* wi   = (float*)carve(NN * sizeof(float));
  float* sbuf = (float*)carve((size_t)BB * 2 * NN * DD * sizeof(float)); // 16 MB
  float* Abuf = (float*)carve((size_t)BB * 2 * NN * DD * sizeof(float)); // 16 MB
  bf16*  Gr   = (bf16*) carve((size_t)BB * DD * NN * sizeof(bf16));      // 4 MB
  bf16*  Gi   = (bf16*) carve((size_t)BB * DD * NN * sizeof(bf16));      // 4 MB

  (void)hipMemsetAsync(sbuf, 0, (size_t)BB * 2 * NN * DD * sizeof(float), stream);

  dss_prep0 <<<2, 256, 0, stream>>>(Lr, Li, lrA, liA, Zr, Zi, wr, wi);
  dss_prepc <<<(DD * NN) / 256, 256, 0, stream>>>(Cr, Ci, wr, wi, Cpr, Cpi, Cbr, Cbi);
  dss_prepse<<<(QQ * NN) / 256, 256, 0, stream>>>(lrA, liA, KSr, KSin, Er, Ein, Pcat);
  dss_ln    <<<BB * LL, 256, 0, stream>>>(x, ln_w, ln_b, pD, out, uB);
  dss_transpose<<<dim3(DD / 32, (unsigned)(BL_ / 32)), dim3(32, 8), 0, stream>>>(uB, uT);
  dss_kcon  <<<32 * 2, 32, 0, stream>>>(Cbr, Cbi, KSr, KSin, KfT);
  dss_toeplitz<<<DD, 128, 0, stream>>>(KfT, uT, out);

  for (int c = 0; c < NCH; ++c) {
    if (c > 0) {
      dss_mkg <<<(BB * DD * NN) / 256, 256, 0, stream>>>(sbuf, Cpr, Cpi, Gr, Gi);
      dss_emit<<<4 * 16 * BB, 32, 0, stream>>>(Er, Ein, Gr, Gi, out, c);
    }
    if (c < NCH - 1) {
      dss_collect<<<32 * 16 * BB, 32, 0, stream>>>(Pcat, uT, Abuf, c);
      dss_update <<<(BB * NN * DD) / 256, 256, 0, stream>>>(sbuf, Abuf, Zr, Zi);
    }
  }
}
